// AttnDecoderRNN_24507083391688
// MI455X (gfx1250) — compile-verified, matrix operand order checked by reading, not run on hardware
//
#include <hip/hip_runtime.h>
#include <hip/hip_bf16.h>

#define V_  32000
#define E_  512
#define H_  1024
#define S_  128
#define G_  4096   // 4*H
#define C2_ 2048   // 2*H

typedef __attribute__((ext_vector_type(16))) __bf16        v16bf;
typedef __attribute__((ext_vector_type(8)))  float         v8f;
typedef __attribute__((ext_vector_type(4)))  unsigned int  u32x4;

union V16 { v16bf v; u32x4 q[2]; };

__device__ __forceinline__ unsigned short f32_to_bf16_bits(float f) {
    unsigned int u = __float_as_uint(f);
    u += 0x7FFFu + ((u >> 16) & 1u);   // round-to-nearest-even
    return (unsigned short)(u >> 16);
}

// ---------------- one-time kernels ----------------
__global__ void k_f32_to_bf16(const float* __restrict__ in,
                              unsigned short* __restrict__ out, long long n) {
    long long i = (long long)blockIdx.x * blockDim.x + threadIdx.x;
    long long stride = (long long)gridDim.x * blockDim.x;
    for (; i < n; i += stride) out[i] = f32_to_bf16_bits(in[i]);
}

__global__ void k_init_state(const float* __restrict__ hn, const float* __restrict__ cn,
                             float* __restrict__ h, float* __restrict__ c,
                             unsigned short* __restrict__ h_bf, int* __restrict__ tok) {
    int j = threadIdx.x;                 // 1024
    float hv = hn[j];
    h[j] = hv;  c[j] = cn[j];
    h_bf[j] = f32_to_bf16_bits(hv);
    if (j == 0) *tok = 2;                // SOS token
}

// ---------------- per-step kernels ----------------
__global__ void k_prep_x(const float* __restrict__ emb, const int* __restrict__ tok,
                         unsigned short* __restrict__ x_bf) {
    int t = threadIdx.x;                 // 512
    int id = *tok;
    x_bf[t] = f32_to_bf16_bits(emb[(long long)id * E_ + t]);
}

// GEMV via WMMA: each wave computes 16 output rows of W(K cols, row-major bf16) x vec.
// A-tile per ISA layout: lane = hi*16+m holds row m, K chunks {hi*8..+7} and {hi*8+16..+7}.
// B: vector replicated into all 16 columns (same loads per half-wave).
__device__ __forceinline__ v8f wmma_gemv(const unsigned short* __restrict__ W,
                                         const unsigned short* __restrict__ x,
                                         int K, int row0, v8f acc) {
    const int lane = threadIdx.x & 31;
    const int m    = lane & 15;
    const int hi   = lane >> 4;
    const unsigned short* wb = W + (long long)(row0 + m) * K + hi * 8;
    const unsigned short* xb = x + hi * 8;
    for (int k0 = 0; k0 < K; k0 += 32) {
        V16 A, B;
        A.q[0] = *(const u32x4*)(wb + k0);
        A.q[1] = *(const u32x4*)(wb + k0 + 16);
        B.q[0] = *(const u32x4*)(xb + k0);
        B.q[1] = *(const u32x4*)(xb + k0 + 16);
        acc = __builtin_amdgcn_wmma_f32_16x16x32_bf16(false, A.v, false, B.v,
                                                      (short)0, acc, false, false);
    }
    return acc;
}

__global__ void k_gates(const unsigned short* __restrict__ Wih,
                        const unsigned short* __restrict__ Whh,
                        const unsigned short* __restrict__ x_bf,
                        const unsigned short* __restrict__ h_bf,
                        const float* __restrict__ b_ih, const float* __restrict__ b_hh,
                        float* __restrict__ gates) {
    int wave = (blockIdx.x * blockDim.x + threadIdx.x) >> 5;
    int row0 = wave << 4;                                   // 256 waves * 16 = 4096 rows
    v8f acc = {};
    acc = wmma_gemv(Wih, x_bf, E_, row0, acc);
    acc = wmma_gemv(Whh, h_bf, H_, row0, acc);
    int lane = threadIdx.x & 31;
    if ((lane & 15) == 0) {                                 // N==0 columns: lanes 0 and 16
        int mb = row0 + (lane >> 4) * 8;
        #pragma unroll
        for (int r = 0; r < 8; ++r)
            gates[mb + r] = acc[r] + b_ih[mb + r] + b_hh[mb + r];
    }
}

__global__ void k_lstm(const float* __restrict__ gates,
                       float* __restrict__ h, float* __restrict__ c,
                       unsigned short* __restrict__ h_bf,
                       unsigned short* __restrict__ cat_bf) {
    int j = threadIdx.x;                 // 1024
    float gi = gates[j], gf = gates[H_ + j], gg = gates[2*H_ + j], go = gates[3*H_ + j];
    float ii = 1.f / (1.f + expf(-gi));
    float ff = 1.f / (1.f + expf(-gf));
    float g  = tanhf(gg);
    float oo = 1.f / (1.f + expf(-go));
    float c2 = ff * c[j] + ii * g;
    float h2 = oo * tanhf(c2);
    c[j] = c2;  h[j] = h2;
    unsigned short hb = f32_to_bf16_bits(h2);
    h_bf[j] = hb;                        // next step's W_hh input
    cat_bf[j] = hb;                      // first half of fc input
}

__global__ void k_scores(const float* __restrict__ enc, const float* __restrict__ h,
                         float* __restrict__ scores) {
    __shared__ float red[256];
    int s = blockIdx.x, tid = threadIdx.x;
    const float* row = enc + (long long)s * H_;
    float p = 0.f;
    for (int k = tid; k < H_; k += 256) p += row[k] * h[k];
    red[tid] = p; __syncthreads();
    for (int off = 128; off > 0; off >>= 1) {
        if (tid < off) red[tid] += red[tid + off];
        __syncthreads();
    }
    if (tid == 0) scores[s] = red[0];
}

__global__ void k_softmax(const float* __restrict__ scores, float* __restrict__ p) {
    __shared__ float red[S_];
    int tid = threadIdx.x;               // 128
    float v = scores[tid];
    red[tid] = v; __syncthreads();
    for (int off = S_/2; off > 0; off >>= 1) {
        if (tid < off) red[tid] = fmaxf(red[tid], red[tid + off]);
        __syncthreads();
    }
    float mx = red[0]; __syncthreads();
    float e = expf(v - mx);
    red[tid] = e; __syncthreads();
    for (int off = S_/2; off > 0; off >>= 1) {
        if (tid < off) red[tid] += red[tid + off];
        __syncthreads();
    }
    p[tid] = e / red[0];
}

__global__ void k_ctx(const float* __restrict__ enc, const float* __restrict__ p,
                      unsigned short* __restrict__ cat_bf) {
    __shared__ float sp[S_];
    int tid = threadIdx.x;               // 128
    int j = blockIdx.x * 128 + tid;      // 0..1023
    sp[tid] = p[tid];
    __syncthreads();
    float acc = 0.f;
    for (int s = 0; s < S_; ++s) acc += sp[s] * enc[(long long)s * H_ + j];
    cat_bf[H_ + j] = f32_to_bf16_bits(acc);
}

__global__ void k_fc(const unsigned short* __restrict__ Wfc,
                     const unsigned short* __restrict__ cat_bf,
                     const float* __restrict__ b_fc, float* __restrict__ logits) {
    int wave = (blockIdx.x * blockDim.x + threadIdx.x) >> 5;
    int row0 = wave << 4;                                   // 2000 waves * 16 = 32000 rows
    v8f acc = {};
    acc = wmma_gemv(Wfc, cat_bf, C2_, row0, acc);
    int lane = threadIdx.x & 31;
    if ((lane & 15) == 0) {
        int mb = row0 + (lane >> 4) * 8;
        #pragma unroll
        for (int r = 0; r < 8; ++r)
            logits[mb + r] = acc[r] + b_fc[mb + r];
    }
}

__global__ void k_reduce(const float* __restrict__ logits,
                         int* __restrict__ tok, float* __restrict__ norm,
                         float* __restrict__ out_idx) {
    __shared__ float sv[1024];
    __shared__ int   si[1024];
    int tid = threadIdx.x;               // 1024
    float best = -__builtin_inff(); int bi = 0;
    for (int k = tid; k < V_; k += 1024) {
        float v = logits[k];
        if (v > best) { best = v; bi = k; }
    }
    sv[tid] = best; si[tid] = bi; __syncthreads();
    for (int off = 512; off > 0; off >>= 1) {
        if (tid < off) {
            float ov = sv[tid + off]; int oi = si[tid + off];
            if (ov > sv[tid] || (ov == sv[tid] && oi < si[tid])) { sv[tid] = ov; si[tid] = oi; }
        }
        __syncthreads();
    }
    float mx = sv[0]; int amax = si[0];
    __syncthreads();
    float s = 0.f;
    for (int k = tid; k < V_; k += 1024) s += expf(logits[k] - mx);
    sv[tid] = s; __syncthreads();
    for (int off = 512; off > 0; off >>= 1) {
        if (tid < off) sv[tid] += sv[tid + off];
        __syncthreads();
    }
    if (tid == 0) {
        *tok = amax;
        *norm = mx + logf(sv[0]);
        *out_idx = (float)amax;
    }
}

__global__ void k_preds(const float* __restrict__ logits, const float* __restrict__ norm,
                        float* __restrict__ out) {
    int gid = blockIdx.x * blockDim.x + threadIdx.x;        // 32000
    out[gid] = logits[gid] - *norm;
}

// ---------------- launcher ----------------
extern "C" void kernel_launch(void* const* d_in, const int* in_sizes, int n_in,
                              void* d_out, int out_size, void* d_ws, size_t ws_size,
                              hipStream_t stream) {
    const float* hn  = (const float*)d_in[0];
    const float* cn  = (const float*)d_in[1];
    const float* enc = (const float*)d_in[2];
    const float* emb = (const float*)d_in[3];
    const float* Wih = (const float*)d_in[4];
    const float* Whh = (const float*)d_in[5];
    const float* bih = (const float*)d_in[6];
    const float* bhh = (const float*)d_in[7];
    const float* Wfc = (const float*)d_in[8];
    const float* bfc = (const float*)d_in[9];
    // cutoff lives on device (can't sync-read during graph capture); recover from out_size:
    // out = cutoff*V preds + cutoff indices
    int cutoff = out_size / (V_ + 1);

    char* p = (char*)d_ws;
    auto carve = [&](size_t bytes) {
        char* r = p; p += (bytes + 255) & ~(size_t)255; return r;
    };
    unsigned short* Wfc_bf = (unsigned short*)carve((size_t)V_ * C2_ * 2);
    unsigned short* Wih_bf = (unsigned short*)carve((size_t)G_ * E_ * 2);
    unsigned short* Whh_bf = (unsigned short*)carve((size_t)G_ * H_ * 2);
    unsigned short* x_bf   = (unsigned short*)carve(E_ * 2);
    unsigned short* h_bf   = (unsigned short*)carve(H_ * 2);
    unsigned short* cat_bf = (unsigned short*)carve(C2_ * 2);
    float* h      = (float*)carve(H_ * 4);
    float* c      = (float*)carve(H_ * 4);
    float* gates  = (float*)carve(G_ * 4);
    float* scores = (float*)carve(S_ * 4);
    float* probs  = (float*)carve(S_ * 4);
    float* logits = (float*)carve(V_ * 4);
    int*   tok    = (int*)carve(4);
    float* norm   = (float*)carve(4);

    float* out = (float*)d_out;
    float* out_idx = out + (size_t)cutoff * V_;

    // One-time bf16 weight conversion (per call; deterministic)
    k_f32_to_bf16<<<2048, 256, 0, stream>>>(Wfc, Wfc_bf, (long long)V_ * C2_);
    k_f32_to_bf16<<<512, 256, 0, stream>>>(Wih, Wih_bf, (long long)G_ * E_);
    k_f32_to_bf16<<<512, 256, 0, stream>>>(Whh, Whh_bf, (long long)G_ * H_);
    k_init_state<<<1, 1024, 0, stream>>>(hn, cn, h, c, h_bf, tok);

    for (int t = 0; t < cutoff; ++t) {
        k_prep_x <<<1, 512, 0, stream>>>(emb, tok, x_bf);
        k_gates  <<<G_ / 128, 256, 0, stream>>>(Wih_bf, Whh_bf, x_bf, h_bf, bih, bhh, gates);
        k_lstm   <<<1, 1024, 0, stream>>>(gates, h, c, h_bf, cat_bf);
        k_scores <<<S_, 256, 0, stream>>>(enc, h, scores);
        k_softmax<<<1, S_, 0, stream>>>(scores, probs);
        k_ctx    <<<H_ / 128, 128, 0, stream>>>(enc, probs, cat_bf);
        k_fc     <<<V_ / 128, 256, 0, stream>>>(Wfc_bf, cat_bf, bfc, logits);
        k_reduce <<<1, 1024, 0, stream>>>(logits, tok, norm, out_idx + t);
        k_preds  <<<V_ / 256, 256, 0, stream>>>(logits, norm, out + (size_t)t * V_);
    }
    (void)in_sizes; (void)n_in; (void)ws_size;
}
